// F_cell_13958643712868
// MI455X (gfx1250) — matrix operand order repaired and checked by reading, not compiled, not run on hardware
//
#include <hip/hip_runtime.h>

#define B_DIM   4096
#define IN_DIM  1024
#define OUT_DIM 2048

// ---- WMMA operand vector types (probe-confirmed builtin signature) ----
typedef __bf16 v16bf __attribute__((ext_vector_type(16)));
typedef __bf16 v8bf  __attribute__((ext_vector_type(8)));
typedef float  v8f   __attribute__((ext_vector_type(8)));

union Pack8 { unsigned short s[8]; uint4 v; };
union Frag  { v16bf v; v8bf h[2]; };

// f32 -> bf16 round-to-nearest-even
__device__ __forceinline__ unsigned short f2bf(float f) {
  union { float f; unsigned int u; } x;
  x.f = f;
  unsigned int r = x.u + 0x7FFFu + ((x.u >> 16) & 1u);
  return (unsigned short)(r >> 16);
}

// Per-lane async 16B copy global -> LDS (gfx1250, tracked by ASYNCcnt).
// Generic shared pointers carry the LDS byte offset in their low 32 bits.
__device__ __forceinline__ void async_copy16(void* lds_dst, const void* gsrc) {
  unsigned loff = (unsigned)(unsigned long long)lds_dst;
  asm volatile("global_load_async_to_lds_b128 %0, %1, off"
               :: "v"(loff), "v"(gsrc)
               : "memory");
}
__device__ __forceinline__ void wait_async0() {
  asm volatile("s_wait_asynccnt 0x0" ::: "memory");
}

// ---------------------------------------------------------------------------
// LDS tile layouts (all bf16, per-block = 16 rows/cols of the 128-wide tile):
//  A block i (16 rows x 32 k):  lA[i*512 + t*256 + lane*8 + e]
//     lane = (m&15) + 16*h ; element e in chunk t holds K = 16*h + 8*t + e
//     (contiguous 16-K per lane half, matching B)
//  B block j (32 k x 16 n):     lB[j*512 + lane*16 + e]
//     lane = (n&15) + 16*h ; element e holds K = 16*h + e
//     (contiguous 16-K per lane half, same order as A)
// ---------------------------------------------------------------------------

// GEMM1: pre = ht@Wh + xt@Wx + b ; r = alpha*ht - relu(pre) ; m = (pre>0)?r:0
__global__ __launch_bounds__(256)
void k1_phi(const float* __restrict__ xt, const float* __restrict__ ht,
            const float* __restrict__ Wh, const float* __restrict__ Wx,
            const float* __restrict__ bias, const float* __restrict__ alphaPtr,
            float* __restrict__ ws_r, unsigned short* __restrict__ ws_m) {
  __shared__ __align__(16) unsigned short lA[2][4096];
  __shared__ __align__(16) unsigned short lB[2][4096];

  const int tid  = threadIdx.x;
  const int lane = tid & 31;
  const int wave = tid >> 5;
  const int wr   = wave >> 1;   // 0..3 : 32-row strip
  const int wc   = wave & 1;    // 0..1 : 64-col strip
  const int bm   = blockIdx.y * 128;
  const int bn   = blockIdx.x * 128;

  v8f acc[2][4];
#pragma unroll
  for (int i = 0; i < 2; ++i)
#pragma unroll
    for (int j = 0; j < 4; ++j)
#pragma unroll
      for (int g = 0; g < 8; ++g) acc[i][j][g] = 0.0f;

  // Stage one 128x32 A tile + 32x128 B tile (f32 -> bf16) into buffer `buf`.
  auto stage = [&](int buf, int s) {
    const float* Asrc; const float* Bsrc; int lda, kbase;
    if (s < 64) { Asrc = ht; lda = OUT_DIM; kbase = s * 32;        Bsrc = Wh; }
    else        { Asrc = xt; lda = IN_DIM;  kbase = (s - 64) * 32; Bsrc = Wx; }
#pragma unroll
    for (int p = 0; p < 2; ++p) {
      int idx = p * 256 + tid;
      int row = idx >> 2;
      int seg = idx & 3;                 // 8 consecutive k = seg*8..seg*8+7
      int h = seg >> 1, t = seg & 1;
      const float4* src =
          (const float4*)(Asrc + (size_t)(bm + row) * lda + kbase + seg * 8);
      float4 f0 = src[0], f1 = src[1];
      Pack8 pk;
      pk.s[0] = f2bf(f0.x); pk.s[1] = f2bf(f0.y);
      pk.s[2] = f2bf(f0.z); pk.s[3] = f2bf(f0.w);
      pk.s[4] = f2bf(f1.x); pk.s[5] = f2bf(f1.y);
      pk.s[6] = f2bf(f1.z); pk.s[7] = f2bf(f1.w);
      *(uint4*)&lA[buf][(row >> 4) * 512 + t * 256 + ((row & 15) + 16 * h) * 8] =
          pk.v;
    }
    {
      const int n  = tid & 127;
      const int kt = tid >> 7;           // lane half h = kt, K = 16*kt + kk
      const float* src = Bsrc + (size_t)(kbase + kt * 16) * OUT_DIM + bn + n;
      Pack8 lo, hi;
#pragma unroll
      for (int kk = 0; kk < 8; ++kk) lo.s[kk] = f2bf(src[(size_t)kk * OUT_DIM]);
#pragma unroll
      for (int kk = 0; kk < 8; ++kk)
        hi.s[kk] = f2bf(src[(size_t)(kk + 8) * OUT_DIM]);
      unsigned short* dst = &lB[buf][(n >> 4) * 512 + ((n & 15) + 16 * kt) * 16];
      *(uint4*)(dst)     = lo.v;
      *(uint4*)(dst + 8) = hi.v;
    }
    __builtin_prefetch(Bsrc + (size_t)(kbase + (tid >> 4)) * OUT_DIM + bn + 32, 0, 3);
  };

  stage(0, 0);
  for (int step = 0; step < 96; ++step) {
    const int cur = step & 1;
    __syncthreads();                       // buffer `cur` ready for all waves
    if (step + 1 < 96) stage(cur ^ 1, step + 1);  // overlap with compute below

    Frag bfrag[4];
#pragma unroll
    for (int j = 0; j < 4; ++j) {
      const v8bf* p = (const v8bf*)&lB[cur][(wc * 4 + j) * 512 + lane * 16];
      bfrag[j].h[0] = p[0];
      bfrag[j].h[1] = p[1];
    }
#pragma unroll
    for (int i = 0; i < 2; ++i) {
      const int ib = wr * 2 + i;
      Frag af;
      af.h[0] = *(const v8bf*)&lA[cur][ib * 512 + lane * 8];
      af.h[1] = *(const v8bf*)&lA[cur][ib * 512 + 256 + lane * 8];
#pragma unroll
      for (int j = 0; j < 4; ++j)
        acc[i][j] = __builtin_amdgcn_wmma_f32_16x16x32_bf16(
            false, af.v, false, bfrag[j].v, (short)0, acc[i][j], false, false);
    }
  }

  // ---- epilogue: bias, relu, r, masked r ----
  const float alpha = alphaPtr[0];
  const int h = lane >> 4, n = lane & 15;
#pragma unroll
  for (int i = 0; i < 2; ++i) {
    const int rowb = bm + (wr * 2 + i) * 16 + 8 * h;
#pragma unroll
    for (int j = 0; j < 4; ++j) {
      const int col = bn + (wc * 4 + j) * 16 + n;
      const float bb = bias[col];
#pragma unroll
      for (int g = 0; g < 8; ++g) {
        const int row = rowb + g;
        const size_t o = (size_t)row * OUT_DIM + col;
        float pre = acc[i][j][g] + bb;
        float phi = pre > 0.0f ? pre : 0.0f;
        float r   = alpha * ht[o] - phi;
        ws_r[o] = r;
        ws_m[o] = f2bf(pre > 0.0f ? r : 0.0f);
      }
    }
  }
}

// GEMM2: out = alpha*r - m @ Wh^T   (B[k][n] = Wh[n][k], transposed in LDS)
__global__ __launch_bounds__(256)
void k2_grad(const unsigned short* __restrict__ ws_m,
             const float* __restrict__ ws_r, const float* __restrict__ Wh,
             const float* __restrict__ alphaPtr, float* __restrict__ out) {
  __shared__ __align__(16) unsigned short lA[2][4096];
  __shared__ __align__(16) unsigned short lB[2][4096];

  const int tid  = threadIdx.x;
  const int lane = tid & 31;
  const int wave = tid >> 5;
  const int wr   = wave >> 1;
  const int wc   = wave & 1;
  const int bm   = blockIdx.y * 128;
  const int bn   = blockIdx.x * 128;

  v8f acc[2][4];
#pragma unroll
  for (int i = 0; i < 2; ++i)
#pragma unroll
    for (int j = 0; j < 4; ++j)
#pragma unroll
      for (int g = 0; g < 8; ++g) acc[i][j][g] = 0.0f;

  // A tile is already bf16 in ws_m: pure byte-move -> per-lane async copy
  // directly into the swizzled fragment layout (ASYNCcnt-tracked).
  auto stageA = [&](int buf, int s) {
    const int k0 = s * 32;
#pragma unroll
    for (int p = 0; p < 2; ++p) {
      int idx = p * 256 + tid;
      int row = idx >> 2;
      int seg = idx & 3;
      int h = seg >> 1, t = seg & 1;
      const unsigned short* g =
          ws_m + (size_t)(bm + row) * OUT_DIM + k0 + seg * 8;
      void* l =
          &lA[buf][(row >> 4) * 512 + t * 256 + ((row & 15) + 16 * h) * 8];
      async_copy16(l, g);
    }
  };
  // B tile = Wh^T with f32->bf16: each thread reads 16 contiguous f32 of a
  // Wh row (64B) and deposits one fragment lane; transpose happens in LDS.
  auto stageB = [&](int buf, int s) {
    const int k0 = s * 32;
    const int n  = tid & 127;          // output column o = bn+n = Wh row
    const int kt = tid >> 7;
    const float4* src =
        (const float4*)(Wh + (size_t)(bn + n) * OUT_DIM + k0 + kt * 16);
    float4 a = src[0], b4 = src[1], c4 = src[2], d4 = src[3];
    Pack8 lo, hi;
    lo.s[0] = f2bf(a.x);  lo.s[1] = f2bf(a.y);
    lo.s[2] = f2bf(a.z);  lo.s[3] = f2bf(a.w);
    lo.s[4] = f2bf(b4.x); lo.s[5] = f2bf(b4.y);
    lo.s[6] = f2bf(b4.z); lo.s[7] = f2bf(b4.w);
    hi.s[0] = f2bf(c4.x); hi.s[1] = f2bf(c4.y);
    hi.s[2] = f2bf(c4.z); hi.s[3] = f2bf(c4.w);
    hi.s[4] = f2bf(d4.x); hi.s[5] = f2bf(d4.y);
    hi.s[6] = f2bf(d4.z); hi.s[7] = f2bf(d4.w);
    unsigned short* dst =
        &lB[buf][(n >> 4) * 512 + ((n & 15) + 16 * kt) * 16];
    *(uint4*)(dst)     = lo.v;
    *(uint4*)(dst + 8) = hi.v;
  };

  stageA(0, 0);
  stageB(0, 0);
  for (int step = 0; step < 64; ++step) {
    const int cur = step & 1;
    wait_async0();        // this wave's async writes into `cur` are complete
    __syncthreads();      // all waves staged `cur`; all done reading `cur^1`
    if (step + 1 < 64) {
      stageA(cur ^ 1, step + 1);   // async: overlaps the WMMAs below
      stageB(cur ^ 1, step + 1);
    }

    Frag bfrag[4];
#pragma unroll
    for (int j = 0; j < 4; ++j) {
      const v8bf* p = (const v8bf*)&lB[cur][(wc * 4 + j) * 512 + lane * 16];
      bfrag[j].h[0] = p[0];
      bfrag[j].h[1] = p[1];
    }
#pragma unroll
    for (int i = 0; i < 2; ++i) {
      const int ib = wr * 2 + i;
      Frag af;
      af.h[0] = *(const v8bf*)&lA[cur][ib * 512 + lane * 8];
      af.h[1] = *(const v8bf*)&lA[cur][ib * 512 + 256 + lane * 8];
#pragma unroll
      for (int j = 0; j < 4; ++j)
        acc[i][j] = __builtin_amdgcn_wmma_f32_16x16x32_bf16(
            false, af.v, false, bfrag[j].v, (short)0, acc[i][j], false, false);
    }
  }

  const float alpha = alphaPtr[0];
  const int h = lane >> 4, n = lane & 15;
#pragma unroll
  for (int i = 0; i < 2; ++i) {
    const int rowb = bm + (wr * 2 + i) * 16 + 8 * h;
#pragma unroll
    for (int j = 0; j < 4; ++j) {
      const int col = bn + (wc * 4 + j) * 16 + n;
#pragma unroll
      for (int g = 0; g < 8; ++g) {
        const size_t o = (size_t)(rowb + g) * OUT_DIM + col;
        out[o] = alpha * ws_r[o] - acc[i][j][g];
      }
    }
  }
}

extern "C" void kernel_launch(void* const* d_in, const int* in_sizes, int n_in,
                              void* d_out, int out_size, void* d_ws, size_t ws_size,
                              hipStream_t stream) {
  const float* xt    = (const float*)d_in[0];
  const float* ht    = (const float*)d_in[1];
  const float* Wh    = (const float*)d_in[2];
  const float* Wx    = (const float*)d_in[3];
  const float* bias  = (const float*)d_in[4];
  const float* alpha = (const float*)d_in[5];
  float* outp = (float*)d_out;

  float* ws_r = (float*)d_ws;                               // 32 MiB
  unsigned short* ws_m =
      (unsigned short*)((char*)d_ws + (size_t)B_DIM * OUT_DIM * sizeof(float));

  dim3 grid(OUT_DIM / 128, B_DIM / 128);  // (16, 32)
  dim3 block(256);
  k1_phi<<<grid, block, 0, stream>>>(xt, ht, Wh, Wx, bias, alpha, ws_r, ws_m);
  k2_grad<<<grid, block, 0, stream>>>(ws_m, ws_r, Wh, alpha, outp);
}
